// ROIBBoxHead_43052752175655
// MI455X (gfx1250) — matrix operand order repaired and checked by reading, not compiled, not run on hardware
//
#include <hip/hip_runtime.h>
#include <hip/hip_bf16.h>

typedef __attribute__((ext_vector_type(16))) __bf16   v16bf;
typedef __attribute__((ext_vector_type(4)))  __bf16   v4bf;
typedef __attribute__((ext_vector_type(8)))  float    v8f;
typedef __attribute__((ext_vector_type(4)))  float    v4f;
typedef __attribute__((ext_vector_type(4)))  int      v4i;
typedef __attribute__((ext_vector_type(8)))  int      v8i;
typedef __attribute__((ext_vector_type(2)))  unsigned v2u;

// fp32 quad -> packed bf16 quad (RNE). Whole-vector bitcast lets the backend
// emit exactly two v_cvt_pk_bf16_f32, with no element extract/reassemble ops.
__device__ __forceinline__ v2u pack4(float a, float b, float c, float d) {
    v4f v = {a, b, c, d};
    v4bf r = __builtin_convertvector(v, v4bf);
    return __builtin_bit_cast(v2u, r);
}

// LDS geometry: K-tile = 64 bf16 = 128B per row, padded to 144B to de-conflict
// banks under 16-lane row-strided ds_load_b128 (m*36 mod 64 distinct for m=0..15).
#define BM   128
#define BN   128
#define BK   64
#define RS   144                 // padded row stride (bytes)
#define ASZ  (BM * RS)           // 18432 B per A buffer
#define BSZ  (BN * RS)           // 18432 B per B buffer (transposed [N][K])
#define STG  (ASZ + BSZ)         // 36864 B per stage

// C[M,N] = act(A[M,K] * B[K,N] + bias), optional ReLU, optional row mask.
// Requires: M % 128 == 0, K % 64 == 0, N % 4 == 0 (N may be non-mult-of-128).
__global__ __launch_bounds__(256)
void roi_mlp_gemm_bf16_wmma(const float* __restrict__ A,
                            const float* __restrict__ B,
                            const float* __restrict__ bias,
                            float* __restrict__ C,
                            int M, int N, int K,
                            int relu, const int* __restrict__ mask)
{
    __shared__ __align__(16) char smem[2 * STG];

    const int tid  = threadIdx.x;
    const int lane = tid & 31;
    const int wave = tid >> 5;
    const int l16  = lane & 15;
    const int half = lane >> 4;            // 0: lanes 0-15, 1: lanes 16-31

    const int m0 = blockIdx.y * BM;
    const int n0 = blockIdx.x * BN;
    const int wm = (wave & 3) * 32;        // wave's 32 rows within block tile
    const int wn = (wave >> 2) * 64;       // wave's 64 cols within block tile

    v8f acc[2][4];
    #pragma unroll
    for (int i = 0; i < 2; ++i)
        #pragma unroll
        for (int j = 0; j < 4; ++j)
            acc[i][j] = v8f{0.f, 0.f, 0.f, 0.f, 0.f, 0.f, 0.f, 0.f};

    // --- staging register tiles -------------------------------------------
    // A: thread covers (mA + 16*i, kA..kA+3), i = 0..7  -> 8 float4
    const int mA = tid >> 4;               // 0..15
    const int kA = (tid & 15) << 2;        // 0,4,...,60
    float4 ra[8];
    // B: slot s = tid + 256*i (i=0..1): k4 = (s>>5)*4, n4 = (s&31)*4
    float4 rb[2][4];

    auto loadA = [&](int k0) {
        #pragma unroll
        for (int i = 0; i < 8; ++i) {
            int m = mA + 16 * i;
            ra[i] = *(const float4*)(A + (size_t)(m0 + m) * K + (k0 + kA));
        }
    };
    auto loadB = [&](int k0) {
        #pragma unroll
        for (int i = 0; i < 2; ++i) {
            int s  = tid + 256 * i;
            int k4 = ((s >> 5) & 15) << 2;
            int n4 = (s & 31) << 2;
            int col = n0 + n4;
            #pragma unroll
            for (int j = 0; j < 4; ++j) {
                if (col < N)
                    rb[i][j] = *(const float4*)(B + (size_t)(k0 + k4 + j) * N + col);
                else
                    rb[i][j] = float4{0.f, 0.f, 0.f, 0.f};
            }
        }
    };
    auto storeA = [&](int st) {
        char* base = smem + st * STG;
        #pragma unroll
        for (int i = 0; i < 8; ++i) {
            int m = mA + 16 * i;
            *(v2u*)(base + m * RS + kA * 2) =
                pack4(ra[i].x, ra[i].y, ra[i].z, ra[i].w);
        }
    };
    auto storeB = [&](int st) {
        char* base = smem + st * STG + ASZ;
        #pragma unroll
        for (int i = 0; i < 2; ++i) {
            int s  = tid + 256 * i;
            int k4 = ((s >> 5) & 15) << 2;
            int n4 = (s & 31) << 2;
            #pragma unroll
            for (int c = 0; c < 4; ++c) {     // transpose: [n][k] layout
                *(v2u*)(base + (n4 + c) * RS + k4 * 2) =
                    pack4(((const float*)&rb[i][0])[c],
                          ((const float*)&rb[i][1])[c],
                          ((const float*)&rb[i][2])[c],
                          ((const float*)&rb[i][3])[c]);
            }
        }
    };

    auto compute = [&](int st) {
        const char* ab = smem + st * STG;
        const char* bb = ab + ASZ;
        #pragma unroll
        for (int kk = 0; kk < 2; ++kk) {      // two 16x16x32 k-steps per stage
            v16bf afrag[2], bfrag[4];
            #pragma unroll
            for (int mt = 0; mt < 2; ++mt) {
                // A layout: lane half 0 -> K 0..15, half 1 -> K 16..31
                const char* p = ab + (wm + mt * 16 + l16) * RS + (half * 16 + kk * 32) * 2;
                v4i lo = *(const v4i*)p;
                v4i hi = *(const v4i*)(p + 16);
                v8i t  = __builtin_shufflevector(lo, hi, 0, 1, 2, 3, 4, 5, 6, 7);
                afrag[mt] = __builtin_bit_cast(v16bf, t);
            }
            #pragma unroll
            for (int nt = 0; nt < 4; ++nt) {
                // B layout ([N][K] in LDS): lane half 0 -> K 0..15, half 1 -> K 16..31
                const char* p = bb + (wn + nt * 16 + l16) * RS + (half * 16 + kk * 32) * 2;
                v4i lo = *(const v4i*)p;
                v4i hi = *(const v4i*)(p + 16);
                v8i t  = __builtin_shufflevector(lo, hi, 0, 1, 2, 3, 4, 5, 6, 7);
                bfrag[nt] = __builtin_bit_cast(v16bf, t);
            }
            #pragma unroll
            for (int mt = 0; mt < 2; ++mt)
                #pragma unroll
                for (int nt = 0; nt < 4; ++nt)
                    acc[mt][nt] = __builtin_amdgcn_wmma_f32_16x16x32_bf16(
                        false, afrag[mt], false, bfrag[nt],
                        (short)0, acc[mt][nt], false, false);
        }
    };

    // --- main double-buffered K loop --------------------------------------
    const int ktiles = K >> 6;
    loadA(0); loadB(0);
    storeA(0); storeB(0);
    __syncthreads();

    int st = 0;
    for (int kt = 0; kt < ktiles; ++kt) {
        const bool more = (kt + 1) < ktiles;
        if (more) { loadA((kt + 1) * BK); loadB((kt + 1) * BK); }
        compute(st);
        __syncthreads();
        if (more) { storeA(st ^ 1); storeB(st ^ 1); }
        __syncthreads();
        st ^= 1;
    }

    // --- epilogue: bias (+ReLU) (+row mask), fp32 store -------------------
    #pragma unroll
    for (int mt = 0; mt < 2; ++mt) {
        #pragma unroll
        for (int nt = 0; nt < 4; ++nt) {
            int col = n0 + wn + nt * 16 + l16;
            if (col >= N) continue;
            float bv = bias[col];
            int rowbase = m0 + wm + mt * 16 + half * 8;
            #pragma unroll
            for (int v = 0; v < 8; ++v) {
                int row = rowbase + v;
                float val = acc[mt][nt][v] + bv;
                if (relu) val = fmaxf(val, 0.f);
                if (mask) val = (mask[row] != 0) ? val : 0.f;
                C[(size_t)row * N + col] = val;
            }
        }
    }
}

extern "C" void kernel_launch(void* const* d_in, const int* in_sizes, int n_in,
                              void* d_out, int out_size, void* d_ws, size_t ws_size,
                              hipStream_t stream) {
    (void)in_sizes; (void)n_in; (void)out_size; (void)ws_size;
    const float* X   = (const float*)d_in[0];   // [8,512,7,7,256] -> [4096,12544]
    const int*   vm  = (const int*)d_in[1];     // [8,512] valid mask
    const float* W1  = (const float*)d_in[3];   // [12544,1024]
    const float* b1  = (const float*)d_in[4];
    const float* W2  = (const float*)d_in[5];   // [1024,1024]
    const float* b2  = (const float*)d_in[6];
    const float* Wp  = (const float*)d_in[7];   // [1024,324]
    const float* bp  = (const float*)d_in[8];
    float* out = (float*)d_out;

    const int M = 4096, H = 1024, K1 = 12544, NC = 324;
    float* H1 = (float*)d_ws;                   // [4096,1024] fp32
    float* H2 = H1 + (size_t)M * H;             // [4096,1024] fp32

    dim3 blk(256);
    roi_mlp_gemm_bf16_wmma<<<dim3(H / 128, M / 128), blk, 0, stream>>>(
        X, W1, b1, H1, M, H, K1, /*relu=*/1, /*mask=*/nullptr);
    roi_mlp_gemm_bf16_wmma<<<dim3(H / 128, M / 128), blk, 0, stream>>>(
        H1, W2, b2, H2, M, H, H, /*relu=*/1, /*mask=*/nullptr);
    roi_mlp_gemm_bf16_wmma<<<dim3((NC + 127) / 128, M / 128), blk, 0, stream>>>(
        H2, Wp, bp, out, M, NC, H, /*relu=*/0, /*mask=*/vm);
}
